// ProtoInduction_30889404793010
// MI455X (gfx1250) — matrix operand order repaired and checked by reading, not compiled, not run on hardware
//
#include <hip/hip_runtime.h>
#include <math.h>

// ---------------------------------------------------------------------------
// ProtoInduction on MI455X (gfx1250): bf16 WMMA pipeline.
// n=100, k=5, q=75, d=4096, NQ=7500. Output: (7500,100) fp32.
// All three GEMMs use v_wmma_f32_16x16x32_bf16 with contiguous fragment loads
// (GEMM2 consumes a one-time LDS-tiled transpose of Qn).
// ---------------------------------------------------------------------------

typedef __bf16 bf16;
typedef __attribute__((ext_vector_type(8)))  __bf16 v8bf;
typedef __attribute__((ext_vector_type(16))) __bf16 v16bf;
typedef __attribute__((ext_vector_type(8)))  float  v8f;

#define N_CLS   100
#define K_SUP   5
#define Q_QRY   75
#define KQ      (K_SUP + Q_QRY)   // 80
#define DIM     4096
#define NQ      (N_CLS * Q_QRY)   // 7500
#define MPAD    112               // 7 tiles of 16 (>=100)
#define KPAD    7520              // 235*32 (>=7500)
#define EPSF    1e-8f

// ---------------- elementwise / reduction kernels --------------------------

// proto[c][j] = mean over 5 support rows of x
__global__ void proto_mean_kernel(const float* __restrict__ x, float* __restrict__ proto) {
    int idx = blockIdx.x * blockDim.x + threadIdx.x;
    if (idx >= N_CLS * DIM) return;
    int c = idx >> 12;           // /4096
    int j = idx & (DIM - 1);
    const float* base = x + (size_t)c * KQ * DIM + j;
    float s = 0.f;
#pragma unroll
    for (int r = 0; r < K_SUP; ++r) s += base[(size_t)r * DIM];
    proto[idx] = s * 0.2f;
}

// one block per query row: compute norm, write normalized bf16 row, save norm
__global__ void normalize_q_kernel(const float* __restrict__ x,
                                   bf16* __restrict__ Qn, float* __restrict__ qnorm) {
    __shared__ float red[256];
    int j = blockIdx.x;                    // query index 0..7499
    int c = j / Q_QRY, qi = j % Q_QRY;
    const float* row = x + ((size_t)c * KQ + K_SUP + qi) * DIM;
    int tid = threadIdx.x;
    float vals[16];
    float ss = 0.f;
#pragma unroll
    for (int i = 0; i < 16; ++i) { vals[i] = row[tid + i * 256]; ss += vals[i] * vals[i]; }
    red[tid] = ss; __syncthreads();
    for (int s = 128; s > 0; s >>= 1) {
        if (tid < s) red[tid] += red[tid + s];
        __syncthreads();
    }
    float nrm = fmaxf(sqrtf(red[0]), EPSF);
    float inv = 1.f / nrm;
    if (tid == 0) qnorm[j] = nrm;
    bf16* orow = Qn + (size_t)j * DIM;
#pragma unroll
    for (int i = 0; i < 16; ++i) orow[tid + i * 256] = (bf16)(vals[i] * inv);
}

// one block per class: normalize proto, save self_sim = sum(proto_n^2)
__global__ void normalize_p_kernel(const float* __restrict__ proto,
                                   bf16* __restrict__ Pn, float* __restrict__ selfsim) {
    __shared__ float red[256];
    int c = blockIdx.x, tid = threadIdx.x;
    const float* row = proto + (size_t)c * DIM;
    float vals[16];
    float ss = 0.f;
#pragma unroll
    for (int i = 0; i < 16; ++i) { vals[i] = row[tid + i * 256]; ss += vals[i] * vals[i]; }
    red[tid] = ss; __syncthreads();
    for (int s = 128; s > 0; s >>= 1) {
        if (tid < s) red[tid] += red[tid + s];
        __syncthreads();
    }
    float sum = red[0];
    float inv = 1.f / fmaxf(sqrtf(sum), EPSF);
    if (tid == 0) selfsim[c] = sum * inv * inv;
    bf16* orow = Pn + (size_t)c * DIM;
#pragma unroll
    for (int i = 0; i < 16; ++i) orow[tid + i * 256] = (bf16)(vals[i] * inv);
}

// LDS-tiled transpose: QnT[n][j] = Qn[j][n], zero-filled for j >= NQ.
// grid = (KPAD/32, DIM/32), block = 256 (32x8).
__global__ void transpose_q_kernel(const bf16* __restrict__ Qn, bf16* __restrict__ QnT) {
    __shared__ bf16 tile[32][33];          // +1 pad avoids bank conflicts
    int jt = blockIdx.x;                   // 0..234 (K tiles)
    int nt = blockIdx.y;                   // 0..127 (D tiles)
    int tx = threadIdx.x & 31;
    int ty = threadIdx.x >> 5;             // 0..7
#pragma unroll
    for (int r = 0; r < 4; ++r) {
        int j = jt * 32 + ty + r * 8;
        bf16 v = (bf16)0.f;
        if (j < NQ) v = Qn[(size_t)j * DIM + nt * 32 + tx];
        tile[ty + r * 8][tx] = v;
    }
    __syncthreads();
#pragma unroll
    for (int r = 0; r < 4; ++r) {
        int n = nt * 32 + ty + r * 8;
        QnT[(size_t)n * KPAD + jt * 32 + tx] = tile[tx][ty + r * 8];
    }
}

// ---------------- WMMA fragment helpers ------------------------------------

// A fragment: lane holds row (lane&15); K = 16*kh .. 16*kh+15 contiguous
__device__ __forceinline__ v16bf load_a_frag(const bf16* __restrict__ A, int row,
                                             int lda, int k0, int khalf) {
    const bf16* p = A + (size_t)row * lda + k0 + khalf * 16;
    v8bf lo = *(const v8bf*)(p);
    v8bf hi = *(const v8bf*)(p + 8);
    v16bf a;
#pragma unroll
    for (int i = 0; i < 8; ++i) { a[i] = lo[i]; a[i + 8] = hi[i]; }
    return a;
}

// B fragment when B-columns are contiguous rows of src: lane holds col (lane&15),
// K = 16*kh .. 16*kh+15 contiguous.
__device__ __forceinline__ v16bf load_b_frag_rows(const bf16* __restrict__ B, int colrow,
                                                  int ldb, int k0, int khalf) {
    return *(const v16bf*)(B + (size_t)colrow * ldb + k0 + khalf * 16);
}

// ---------------- GEMM 1 & 3: (7500 x 4096) @ (4096 x Ncls^T) --------------
// Out[M*100+N] = alpha * sum_k Qn[M][k] * Prow[N][k]
__global__ void gemm_qp_kernel(const bf16* __restrict__ Qn, const bf16* __restrict__ Prow,
                               float* __restrict__ Out, const float* __restrict__ taoPtr,
                               int applyTao) {
    const int MT = (NQ + 15) / 16;      // 469
    const int NT = (N_CLS + 15) / 16;   // 7
    int wave = threadIdx.x >> 5;
    int lane = threadIdx.x & 31;
    int t = blockIdx.x * 8 + wave;
    if (t >= MT * NT) return;
    int mt = t / NT, nt = t % NT;
    int khalf = lane >> 4;
    int arow = mt * 16 + (lane & 15); if (arow > NQ - 1) arow = NQ - 1;     // clamp (discarded rows)
    int brow = nt * 16 + (lane & 15); if (brow > N_CLS - 1) brow = N_CLS - 1;

    v8f acc = {};
    for (int k0 = 0; k0 < DIM; k0 += 32) {
        v16bf a = load_a_frag(Qn, arow, DIM, k0, khalf);
        v16bf b = load_b_frag_rows(Prow, brow, DIM, k0, khalf);
        acc = __builtin_amdgcn_wmma_f32_16x16x32_bf16(false, a, false, b,
                                                      (short)0, acc, false, false);
    }
    float alpha = applyTao ? taoPtr[0] : 1.0f;
    int Nc = nt * 16 + (lane & 15);
#pragma unroll
    for (int r = 0; r < 8; ++r) {
        int M = mt * 16 + r + 8 * khalf;
        if (M < NQ && Nc < N_CLS) Out[(size_t)M * N_CLS + Nc] = alpha * acc[r];
    }
}

// ---------------- argmax over classes per query ----------------------------
__global__ void argmax_kernel(const float* __restrict__ pre_sim, int* __restrict__ label) {
    int j = blockIdx.x * blockDim.x + threadIdx.x;
    if (j >= NQ) return;
    const float* row = pre_sim + (size_t)j * N_CLS;
    float best = row[0]; int bi = 0;
    for (int c = 1; c < N_CLS; ++c) {
        float v = row[c];
        if (v > best) { best = v; bi = c; }   // first max wins (strict >)
    }
    label[j] = bi;
}

// ---------------- masked softmax stats per class ---------------------------
__global__ void stats_kernel(const float* __restrict__ pre_sim, const int* __restrict__ label,
                             const float* __restrict__ selfsim,
                             float* __restrict__ mstat, float* __restrict__ sstat,
                             float* __restrict__ wself) {
    __shared__ float red[256];
    int c = blockIdx.x, tid = threadIdx.x;
    float m = -__builtin_inff();
    for (int j = tid; j < NQ; j += 256)
        if (label[j] == c) m = fmaxf(m, pre_sim[(size_t)j * N_CLS + c]);
    red[tid] = m; __syncthreads();
    for (int s = 128; s > 0; s >>= 1) {
        if (tid < s) red[tid] = fmaxf(red[tid], red[tid + s]);
        __syncthreads();
    }
    float mx = fmaxf(red[0], selfsim[c]);
    __syncthreads();
    float sm = 0.f;
    for (int j = tid; j < NQ; j += 256)
        if (label[j] == c) sm += expf(pre_sim[(size_t)j * N_CLS + c] - mx);
    red[tid] = sm; __syncthreads();
    for (int s = 128; s > 0; s >>= 1) {
        if (tid < s) red[tid] += red[tid + s];
        __syncthreads();
    }
    float eself = expf(selfsim[c] - mx);
    float total = red[0] + eself;
    if (tid == 0) { mstat[c] = mx; sstat[c] = total; wself[c] = eself / total; }
}

// ---------------- materialize W (112 x 7520, bf16, zero-padded) ------------
// W[c][j] = (label[j]==c) ? exp(pre_sim[j][c]-m[c])/s[c] * ||q_j|| : 0
__global__ void weights_kernel(const float* __restrict__ pre_sim, const int* __restrict__ label,
                               const float* __restrict__ mstat, const float* __restrict__ sstat,
                               const float* __restrict__ qnorm, bf16* __restrict__ W) {
    int idx = blockIdx.x * blockDim.x + threadIdx.x;
    if (idx >= MPAD * KPAD) return;
    int c = idx / KPAD, j = idx % KPAD;
    float v = 0.f;
    if (c < N_CLS && j < NQ && label[j] == c)
        v = expf(pre_sim[(size_t)j * N_CLS + c] - mstat[c]) / sstat[c] * qnorm[j];
    W[idx] = (bf16)v;
}

// ---------------- GEMM 2: adapted(100x4096) = W(100x7520) @ Qn(7500x4096) --
// B columns come from the transposed copy QnT (rows contiguous over K).
__global__ void gemm_wq_kernel(const bf16* __restrict__ W, const bf16* __restrict__ QnT,
                               float* __restrict__ adapted) {
    const int MT = MPAD / 16;    // 7
    const int NT = DIM / 16;     // 256
    int wave = threadIdx.x >> 5;
    int lane = threadIdx.x & 31;
    int t = blockIdx.x * 8 + wave;
    if (t >= MT * NT) return;
    int mt = t / NT, nt = t % NT;
    int khalf = lane >> 4;
    int arow = mt * 16 + (lane & 15);          // < 112, W is padded: always valid
    int ncol = nt * 16 + (lane & 15);          // < 4096 always

    v8f acc = {};
    for (int k0 = 0; k0 < KPAD; k0 += 32) {
        v16bf a = load_a_frag(W, arow, KPAD, k0, khalf);
        v16bf b = load_b_frag_rows(QnT, ncol, KPAD, k0, khalf);   // contiguous 32B
        acc = __builtin_amdgcn_wmma_f32_16x16x32_bf16(false, a, false, b,
                                                      (short)0, acc, false, false);
    }
#pragma unroll
    for (int r = 0; r < 8; ++r) {
        int M = mt * 16 + r + 8 * khalf;
        if (M < N_CLS) adapted[(size_t)M * DIM + ncol] = acc[r];
    }
}

// ---------------- finish adapted proto: + wself*proto, normalize, bf16 -----
__global__ void adapt_finish_kernel(const float* __restrict__ adapted,
                                    const float* __restrict__ proto,
                                    const float* __restrict__ wself,
                                    bf16* __restrict__ Apn) {
    __shared__ float red[256];
    int c = blockIdx.x, tid = threadIdx.x;
    float wsf = wself[c];
    const float* arow = adapted + (size_t)c * DIM;
    const float* prow = proto + (size_t)c * DIM;
    float vals[16];
    float ss = 0.f;
#pragma unroll
    for (int i = 0; i < 16; ++i) {
        float v = arow[tid + i * 256] + wsf * prow[tid + i * 256];
        vals[i] = v; ss += v * v;
    }
    red[tid] = ss; __syncthreads();
    for (int s = 128; s > 0; s >>= 1) {
        if (tid < s) red[tid] += red[tid + s];
        __syncthreads();
    }
    float inv = 1.f / fmaxf(sqrtf(red[0]), EPSF);
    bf16* orow = Apn + (size_t)c * DIM;
#pragma unroll
    for (int i = 0; i < 16; ++i) orow[tid + i * 256] = (bf16)(vals[i] * inv);
}

// ---------------------------------------------------------------------------

extern "C" void kernel_launch(void* const* d_in, const int* in_sizes, int n_in,
                              void* d_out, int out_size, void* d_ws, size_t ws_size,
                              hipStream_t stream) {
    const float* x   = (const float*)d_in[0];
    const float* tao = (const float*)d_in[1];
    float* out = (float*)d_out;

    // workspace carve-up (256B aligned)
    char* ws = (char*)d_ws;
    size_t off = 0;
    auto alloc = [&](size_t bytes) -> char* {
        char* p = ws + off;
        off = (off + bytes + 255) & ~(size_t)255;
        return p;
    };
    float* proto   = (float*)alloc((size_t)N_CLS * DIM * 4);
    bf16*  Qn      = (bf16*) alloc((size_t)NQ * DIM * 2);
    bf16*  QnT     = (bf16*) alloc((size_t)DIM * KPAD * 2);
    bf16*  Pn      = (bf16*) alloc((size_t)N_CLS * DIM * 2);
    float* qnorm   = (float*)alloc((size_t)NQ * 4);
    float* selfsim = (float*)alloc((size_t)N_CLS * 4);
    float* pre_sim = (float*)alloc((size_t)NQ * N_CLS * 4);
    int*   label   = (int*)  alloc((size_t)NQ * 4);
    float* mstat   = (float*)alloc((size_t)N_CLS * 4);
    float* sstat   = (float*)alloc((size_t)N_CLS * 4);
    float* wself   = (float*)alloc((size_t)N_CLS * 4);
    bf16*  W       = (bf16*) alloc((size_t)MPAD * KPAD * 2);
    float* adapted = (float*)alloc((size_t)N_CLS * DIM * 4);
    bf16*  Apn     = (bf16*) alloc((size_t)N_CLS * DIM * 2);

    // 1. prototypes
    proto_mean_kernel<<<(N_CLS * DIM + 255) / 256, 256, 0, stream>>>(x, proto);
    // 2. normalize queries / protos to bf16
    normalize_q_kernel<<<NQ, 256, 0, stream>>>(x, Qn, qnorm);
    normalize_p_kernel<<<N_CLS, 256, 0, stream>>>(proto, Pn, selfsim);
    // 2b. one-time transpose for GEMM2's B operand
    {
        dim3 tg(KPAD / 32, DIM / 32);
        transpose_q_kernel<<<tg, 256, 0, stream>>>(Qn, QnT);
    }
    // 3. pre_sim = Qn @ Pn^T  (WMMA bf16)
    {
        int tiles = ((NQ + 15) / 16) * ((N_CLS + 15) / 16);
        gemm_qp_kernel<<<(tiles + 7) / 8, 256, 0, stream>>>(Qn, Pn, pre_sim, tao, 0);
    }
    // 4. pre_label
    argmax_kernel<<<(NQ + 255) / 256, 256, 0, stream>>>(pre_sim, label);
    // 5. masked softmax stats per class
    stats_kernel<<<N_CLS, 256, 0, stream>>>(pre_sim, label, selfsim, mstat, sstat, wself);
    // 6. weight matrix (includes ||q|| un-normalization factor)
    weights_kernel<<<(MPAD * KPAD + 255) / 256, 256, 0, stream>>>(pre_sim, label, mstat,
                                                                  sstat, qnorm, W);
    // 7. adapted = W @ Qn  (WMMA bf16, contiguous B via QnT)
    {
        int tiles = (MPAD / 16) * (DIM / 16);
        gemm_wq_kernel<<<(tiles + 7) / 8, 256, 0, stream>>>(W, QnT, adapted);
    }
    // 8. adapted += wself*proto; normalize -> bf16
    adapt_finish_kernel<<<N_CLS, 256, 0, stream>>>(adapted, proto, wself, Apn);
    // 9. out = tao * (Qn @ Apn^T)  (WMMA bf16)
    {
        int tiles = ((NQ + 15) / 16) * ((N_CLS + 15) / 16);
        gemm_qp_kernel<<<(tiles + 7) / 8, 256, 0, stream>>>(Qn, Apn, out, tao, 1);
    }
}
